// OdysseyGPT_81956565943136
// MI455X (gfx1250) — matrix operand order repaired and checked by reading, not compiled, not run on hardware
//
#include <hip/hip_runtime.h>
#include <math.h>

#define EMBED 1024
#define NH    16
#define HD    64
#define NL    8
#define BATCH 4
#define SEQT  1024
#define VOCAB 32000
#define NT    (BATCH * SEQT)   // 4096 token rows

typedef __attribute__((ext_vector_type(16))) __bf16 bf16x16;
typedef __attribute__((ext_vector_type(8)))  float  f32x8;

// ---------------------------------------------------------------- helpers ---

static __device__ __forceinline__ __bf16 cvt_bf16(float x) {
  union { float f; unsigned u; } c; c.f = x;
  unsigned r = c.u + 0x7fffu + ((c.u >> 16) & 1u);   // round-to-nearest-even
  union { unsigned short s; __bf16 b; } o; o.s = (unsigned short)(r >> 16);
  return o.b;
}

static __device__ __forceinline__ f32x8 wmma_bf16(bf16x16 a, bf16x16 b, f32x8 c) {
  // (neg_a, A, neg_b, B, c_mod, C, reuse_a, reuse_b)
  return __builtin_amdgcn_wmma_f32_16x16x32_bf16(false, a, false, b, (short)0, c,
                                                 false, false);
}

// Generic pointer to an LDS object: low 32 bits are the LDS byte offset
// (ISA 10.2: LDS aperture maps addr[31:0] directly).
static __device__ __forceinline__ unsigned lds_addr32(const void* p) {
  return (unsigned)(size_t)p;
}

// Async per-lane 16B copy global -> LDS (GLOBAL_LOAD_ASYNC_TO_LDS_B128,
// tracked with ASYNCcnt).
static __device__ __forceinline__ void async_copy_b128(unsigned lds, const void* g) {
  asm volatile("global_load_async_to_lds_b128 %0, %1, off"
               :: "v"(lds), "v"((unsigned long long)(size_t)g) : "memory");
}
static __device__ __forceinline__ void wait_async0() {
  asm volatile("s_wait_asynccnt 0x0" ::: "memory");
}

// A fragment from a bf16 tile: 16(M) x 32(K), row-major, leading dim ld.
// lane<16: M=lane, K={0..7,16..23}; lane>=16: K={8..15,24..31}
static __device__ __forceinline__ bf16x16 load_a_frag(const __bf16* p, int ld) {
  int lane = threadIdx.x & 31;
  int row  = lane & 15;
  int hi   = lane >> 4;
  const __bf16* q = p + row * ld + hi * 8;
  bf16x16 r;
#pragma unroll
  for (int i = 0; i < 8; i++) r[i]     = q[i];
#pragma unroll
  for (int i = 0; i < 8; i++) r[8 + i] = q[16 + i];
  return r;
}

// Same A fragment but sourced from an f32 tile in LDS (converted in-regs;
// the cvt VALU co-executes with WMMA).
static __device__ __forceinline__ bf16x16 load_a_frag_f32(const float* p, int ld) {
  int lane = threadIdx.x & 31;
  int row  = lane & 15;
  int hi   = lane >> 4;
  const float* q = p + row * ld + hi * 8;
  bf16x16 r;
#pragma unroll
  for (int i = 0; i < 8; i++) r[i]     = cvt_bf16(q[i]);
#pragma unroll
  for (int i = 0; i < 8; i++) r[8 + i] = cvt_bf16(q[16 + i]);
  return r;
}

// B fragment: 32(K) x 16(N) bf16 tile stored N-major: element (k,n) at p[n*ld+k].
static __device__ __forceinline__ bf16x16 load_b_frag(const __bf16* p, int ld) {
  int lane = threadIdx.x & 31;
  int n    = lane & 15;
  int hi   = lane >> 4;
  const __bf16* q = p + n * ld + hi * 8;
  bf16x16 r;
#pragma unroll
  for (int i = 0; i < 8; i++) r[i]     = q[i];
#pragma unroll
  for (int i = 0; i < 8; i++) r[8 + i] = q[16 + i];
  return r;
}

// ------------------------------------------------------------------ embed ---

__global__ __launch_bounds__(256) void embed_kernel(
    const float* __restrict__ tok, const float* __restrict__ pos,
    const int* __restrict__ x, float* __restrict__ h) {
  int row = blockIdx.x;                 // 0..NT-1
  int t   = row & (SEQT - 1);
  int id  = x[row];
  const float* te = tok + (size_t)id * EMBED;
  const float* pe = pos + (size_t)t  * EMBED;
  float* hr = h + (size_t)row * EMBED;
  for (int e = threadIdx.x; e < EMBED; e += 256) hr[e] = te[e] + pe[e];
}

// ------------------------------------------------------------------- gemm ---
// C[M,N] = act( A[M,K] @ B + bias ).  B element (k, n) fetched at
//   Bsrc[ (n>>gShift)*gStride + k*kStride + (n & gMask) ]
// covering row-major [K,N] (gShift=31) and the Wq (h,e,d) permutation.
// A tile is staged raw f32 via GLOBAL_LOAD_ASYNC_TO_LDS_B128.

template <int HAS_BIAS, int RELU>
__global__ __launch_bounds__(256) void gemm_bf16_wmma(
    const float* __restrict__ A, const float* __restrict__ Bsrc,
    const float* __restrict__ bias, float* __restrict__ C,
    int M, int N, int K,
    long kStride, int gShift, long gStride, int gMask) {
  __shared__ float  sAf[128][36];    // 128 x 32 f32 (+pad, rows 16B aligned)
  __shared__ __bf16 sB[128][40];     // N-major: [n][k], bf16

  const int tid  = threadIdx.x;
  const int wave = tid >> 5, lane = tid & 31;
  const int wm = wave >> 2, wn = wave & 3;            // 2 x 4 wave grid
  const int m0 = blockIdx.y * 128, n0 = blockIdx.x * 128;
  const int hi = lane >> 4, cl = lane & 15;

  float bcol[2] = {0.f, 0.f};
  if (HAS_BIAS) {
#pragma unroll
    for (int nf = 0; nf < 2; nf++) bcol[nf] = bias[n0 + wn * 32 + nf * 16 + cl];
  }

  const f32x8 zero = {0.f, 0.f, 0.f, 0.f, 0.f, 0.f, 0.f, 0.f};
  f32x8 acc[4][2];
#pragma unroll
  for (int i = 0; i < 4; i++)
#pragma unroll
    for (int j = 0; j < 2; j++) acc[i][j] = zero;

  for (int k0 = 0; k0 < K; k0 += 32) {
    {  // async-stage A tile (raw fp32, 4 x b128 per thread)
#pragma unroll
      for (int c = 0; c < 4; c++) {
        int chunk = c * 256 + tid;          // 0..1023 chunks of 16B
        int row = chunk >> 3;
        int off = (chunk & 7) * 4;
        const float* gp = A + (size_t)(m0 + row) * K + k0 + off;
        async_copy_b128(lds_addr32(&sAf[row][off]), gp);
      }
    }
    {  // stage B tile transposed into N-major bf16 layout
      int kk = tid >> 3, nb = (tid & 7) * 16;
#pragma unroll
      for (int i = 0; i < 16; i++) {
        int gn = n0 + nb + i;
        long idx = ((long)(gn >> gShift)) * gStride +
                   (long)(k0 + kk) * kStride + (long)(gn & gMask);
        sB[nb + i][kk] = cvt_bf16(Bsrc[idx]);
        if (k0 + 32 < K)   // prefetch next B tile row into GL2
          __builtin_prefetch(&Bsrc[idx + 32 * kStride], 0, 1);
      }
    }
    wait_async0();
    __syncthreads();

    bf16x16 aF[4];
#pragma unroll
    for (int mf = 0; mf < 4; mf++)
      aF[mf] = load_a_frag_f32(&sAf[wm * 64 + mf * 16][0], 36);
#pragma unroll
    for (int nf = 0; nf < 2; nf++) {
      bf16x16 bF = load_b_frag(&sB[wn * 32 + nf * 16][0], 40);
#pragma unroll
      for (int mf = 0; mf < 4; mf++) acc[mf][nf] = wmma_bf16(aF[mf], bF, acc[mf][nf]);
    }
    __syncthreads();
  }

#pragma unroll
  for (int mf = 0; mf < 4; mf++) {
    const int row0 = m0 + wm * 64 + mf * 16 + hi * 8;
    float* crow = C + (size_t)row0 * N;
#pragma unroll
    for (int nf = 0; nf < 2; nf++) {
      const int cn = n0 + wn * 32 + nf * 16 + cl;
#pragma unroll
      for (int r = 0; r < 8; r++) {
        float v = acc[mf][nf][r];
        if (HAS_BIAS) v += bcol[nf];
        if (RELU)     v = fmaxf(v, 0.f);
        crow[(size_t)r * N + cn] = v;
      }
    }
  }
}

// -------------------------------------------------------------- attention ---
// Causal flash attention per (b, h, 64-row tile). q serves as q, k and v
// (faithful to the reference). Mask semantics: tril -> 0, then (==0) -> -inf.

#define NEGBIG (-1.0e30f)

__global__ __launch_bounds__(128) void attn_wmma(const float* __restrict__ q,
                                                 float* __restrict__ o) {
  __shared__ __bf16 sQ[64][72];       // 64 rows x 64 dims
  __shared__ __bf16 sK[32][72];       // [key][dim]  (B frag for Q.K^T)
  __shared__ __bf16 sVT[64][40];      // [dim][key]  (B frag for P.V)
  __shared__ __bf16 sP[4][16][40];    // per-wave P tile 16 x 32

  const int tid  = threadIdx.x;
  const int wave = tid >> 5, lane = tid & 31;
  const int hi = lane >> 4, cl = lane & 15;
  const int rt = blockIdx.x, h = blockIdx.y, b = blockIdx.z;
  const int rowBase = rt * 64;
  const float* qb = q + (size_t)b * SEQT * EMBED + (size_t)h * HD;

  {  // stage Q (64 x 64)
    int r = tid >> 1, dh = (tid & 1) * 32;
    const float* g = qb + (size_t)(rowBase + r) * EMBED + dh;
#pragma unroll
    for (int i = 0; i < 32; i++) sQ[r][dh + i] = cvt_bf16(g[i]);
  }
  __syncthreads();

  bf16x16 aQ[2];
  aQ[0] = load_a_frag(&sQ[wave * 16][0], 72);
  aQ[1] = load_a_frag(&sQ[wave * 16][32], 72);

  const f32x8 zero = {0.f, 0.f, 0.f, 0.f, 0.f, 0.f, 0.f, 0.f};
  f32x8 oacc[4];
#pragma unroll
  for (int c = 0; c < 4; c++) oacc[c] = zero;
  float mrow[8], lrow[8];
#pragma unroll
  for (int r = 0; r < 8; r++) { mrow[r] = NEGBIG; lrow[r] = 0.f; }

  const int ntiles = rt * 2 + 2;      // cover keys 0 .. rowBase+63
  for (int jt = 0; jt < ntiles; jt++) {
    const int colBase = jt * 32;
    {  // stage K / V^T (32 keys x 64 dims)
      int kr = tid >> 2, dh = (tid & 3) * 16;
      const float* g = qb + (size_t)(colBase + kr) * EMBED + dh;
      __builtin_prefetch(g + 32 * EMBED, 0, 1);   // next key tile (speculative)
#pragma unroll
      for (int i = 0; i < 16; i++) {
        __bf16 v = cvt_bf16(g[i]);
        sK[kr][dh + i]  = v;
        sVT[dh + i][kr] = v;
      }
    }
    __syncthreads();

    float ps[2][8];
#pragma unroll
    for (int sub = 0; sub < 2; sub++) {
      f32x8 s = zero;
      bf16x16 b0 = load_b_frag(&sK[sub * 16][0], 72);
      s = wmma_bf16(aQ[0], b0, s);
      bf16x16 b1 = load_b_frag(&sK[sub * 16][32], 72);
      s = wmma_bf16(aQ[1], b1, s);
#pragma unroll
      for (int r = 0; r < 8; r++) {
        int rowg = rowBase + wave * 16 + r + hi * 8;
        int colg = colBase + sub * 16 + cl;
        float sv = (colg <= rowg) ? s[r] * 0.125f : 0.0f;  // hd^-0.5 = 1/8
        if (sv == 0.0f) sv = NEGBIG;                        // masked_fill(==0,-inf)
        ps[sub][r] = sv;
      }
    }
    // online softmax: each row lives across 16 lanes (half-wave reductions)
#pragma unroll
    for (int r = 0; r < 8; r++) {
      float tmax = fmaxf(ps[0][r], ps[1][r]);
#pragma unroll
      for (int off = 1; off < 16; off <<= 1)
        tmax = fmaxf(tmax, __shfl_xor(tmax, off, 16));
      float mnew = fmaxf(mrow[r], tmax);
      float f  = __expf(mrow[r] - mnew);
      float p0 = __expf(ps[0][r] - mnew);
      float p1 = __expf(ps[1][r] - mnew);
      float psum = p0 + p1;
#pragma unroll
      for (int off = 1; off < 16; off <<= 1)
        psum += __shfl_xor(psum, off, 16);
      lrow[r] = lrow[r] * f + psum;
      mrow[r] = mnew;
#pragma unroll
      for (int c = 0; c < 4; c++) oacc[c][r] *= f;
      int prow = r + hi * 8;
      sP[wave][prow][cl]      = cvt_bf16(p0);
      sP[wave][prow][16 + cl] = cvt_bf16(p1);
    }
    __syncthreads();

    bf16x16 aP = load_a_frag(&sP[wave][0][0], 40);
#pragma unroll
    for (int c = 0; c < 4; c++) {
      bf16x16 bV = load_b_frag(&sVT[c * 16][0], 40);
      oacc[c] = wmma_bf16(aP, bV, oacc[c]);
    }
    __syncthreads();
  }

  float* ob = o + (size_t)b * SEQT * EMBED + (size_t)h * HD;
#pragma unroll
  for (int c = 0; c < 4; c++)
#pragma unroll
    for (int r = 0; r < 8; r++) {
      int row = rowBase + wave * 16 + r + hi * 8;
      int dim = c * 16 + cl;
      ob[(size_t)row * EMBED + dim] = oacc[c][r] / lrow[r];
    }
}

// -------------------------------------------------------------- layernorm ---
// out = LN(x + res) * g + b   (res may be null); one block per token row.

__global__ __launch_bounds__(256) void ln_kernel(
    const float* __restrict__ x, const float* __restrict__ res,
    const float* __restrict__ g, const float* __restrict__ bt,
    float* __restrict__ out) {
  __shared__ float sb[256];
  const int row = blockIdx.x, tid = threadIdx.x;
  const size_t base = (size_t)row * EMBED;
  float v[4];
  float s = 0.f;
#pragma unroll
  for (int i = 0; i < 4; i++) {
    int e = tid + i * 256;
    float a = x[base + e];
    if (res) a += res[base + e];
    v[i] = a; s += a;
  }
  sb[tid] = s; __syncthreads();
  for (int off = 128; off > 0; off >>= 1) {
    if (tid < off) sb[tid] += sb[tid + off];
    __syncthreads();
  }
  float mu = sb[0] * (1.0f / EMBED);
  __syncthreads();
  s = 0.f;
#pragma unroll
  for (int i = 0; i < 4; i++) { float d = v[i] - mu; s += d * d; }
  sb[tid] = s; __syncthreads();
  for (int off = 128; off > 0; off >>= 1) {
    if (tid < off) sb[tid] += sb[tid + off];
    __syncthreads();
  }
  float rstd = rsqrtf(sb[0] * (1.0f / EMBED) + 1e-5f);
#pragma unroll
  for (int i = 0; i < 4; i++) {
    int e = tid + i * 256;
    out[base + e] = (v[i] - mu) * rstd * g[e] + bt[e];
  }
}

// ------------------------------------------------------------ orchestration -

extern "C" void kernel_launch(void* const* d_in, const int* in_sizes, int n_in,
                              void* d_out, int out_size, void* d_ws, size_t ws_size,
                              hipStream_t stream) {
  const float* tok_emb = (const float*)d_in[0];
  const float* pos_emb = (const float*)d_in[1];
  const float* Wq = (const float*)d_in[2];
  const float* Wo = (const float*)d_in[3];
  const float* bo = (const float*)d_in[4];
  const float* W1 = (const float*)d_in[5];
  const float* b1 = (const float*)d_in[6];
  const float* W2 = (const float*)d_in[7];
  const float* b2 = (const float*)d_in[8];
  const float* ln1_g = (const float*)d_in[9];
  const float* ln1_b = (const float*)d_in[10];
  const float* ln2_g = (const float*)d_in[11];
  const float* ln2_b = (const float*)d_in[12];
  const float* lnf_g = (const float*)d_in[13];
  const float* lnf_b = (const float*)d_in[14];
  const float* Wf = (const float*)d_in[15];
  const float* bf = (const float*)d_in[16];
  const int*   x  = (const int*)d_in[17];
  float* logits = (float*)d_out;

  // workspace: 3 activation buffers of 4096 x 1024 f32 (48 MB total)
  float* hbuf = (float*)d_ws;
  float* t1 = hbuf + (size_t)NT * EMBED;
  float* t2 = t1   + (size_t)NT * EMBED;

  embed_kernel<<<NT, 256, 0, stream>>>(tok_emb, pos_emb, x, hbuf);

  const long EE = (long)EMBED * EMBED;
  dim3 gEE(EMBED / 128, NT / 128);   // (8, 32)
  for (int l = 0; l < NL; l++) {
    // q = h @ Wq[l]   (column n -> head n>>6, dim n&63)
    gemm_bf16_wmma<0, 0><<<gEE, 256, 0, stream>>>(hbuf, Wq + (size_t)l * EE,
        nullptr, t1,
        NT, EMBED, EMBED, /*kStride*/ HD, /*gShift*/ 6,
        /*gStride*/ (long)EMBED * HD, /*gMask*/ HD - 1);
    attn_wmma<<<dim3(SEQT / 64, NH, BATCH), 128, 0, stream>>>(t1, t2);
    gemm_bf16_wmma<1, 0><<<gEE, 256, 0, stream>>>(t2, Wo + (size_t)l * EE,
        bo + (size_t)l * EMBED, t1,
        NT, EMBED, EMBED, EMBED, 31, 0, 0x7fffffff);
    ln_kernel<<<NT, 256, 0, stream>>>(hbuf, t1, ln1_g + (size_t)l * EMBED,
                                      ln1_b + (size_t)l * EMBED, hbuf);
    gemm_bf16_wmma<1, 1><<<gEE, 256, 0, stream>>>(hbuf, W1 + (size_t)l * EE,
        b1 + (size_t)l * EMBED, t1,
        NT, EMBED, EMBED, EMBED, 31, 0, 0x7fffffff);
    gemm_bf16_wmma<1, 0><<<gEE, 256, 0, stream>>>(t1, W2 + (size_t)l * EE,
        b2 + (size_t)l * EMBED, t2,
        NT, EMBED, EMBED, EMBED, 31, 0, 0x7fffffff);
    ln_kernel<<<NT, 256, 0, stream>>>(hbuf, t2, ln2_g + (size_t)l * EMBED,
                                      ln2_b + (size_t)l * EMBED, hbuf);
  }
  ln_kernel<<<NT, 256, 0, stream>>>(hbuf, nullptr, lnf_g, lnf_b, hbuf);
  gemm_bf16_wmma<1, 0><<<dim3(VOCAB / 128, NT / 128), 256, 0, stream>>>(
      hbuf, Wf, bf, logits, NT, VOCAB, EMBED, VOCAB, 31, 0, 0x7fffffff);
}
